// RWKV_RNN_36773509988783
// MI455X (gfx1250) — compile-verified, matrix operand order checked by reading, not run on hardware
//
#include <hip/hip_runtime.h>

// ---------------------------------------------------------------------------
// RWKV-v4 forward (state update only) for gfx1250 / MI455X.
// B=8, T=512, E=1024, H=4096, L=6.  GEMMs use v_wmma_f32_16x16x32_bf16 with
// double-buffered LDS (padded stride 40 elems; conflict-reduced), register-
// transposed f32->bf16 weight staging with v_perm_b32 packing (no spills:
// staging lives in named float4/uint4 registers).  LayerNorm / WKV / epilogues
// fp32.  All 30 state rows of d_out written exactly once per call.  ~180 MB ws.
// ---------------------------------------------------------------------------

typedef __attribute__((ext_vector_type(16))) __bf16 v16bf;
typedef __attribute__((ext_vector_type(8)))  __bf16 v8bf;
typedef __attribute__((ext_vector_type(8)))  float  v8f;

#define RWKV_E 1024
#define BM 128
#define BN 128
#define BK 32
#define TSTR 40   // padded LDS tile row stride in elements (80 B, keeps b128 align)

// Pack two floats to two bf16 (round-half-up, within 1/2 ulp of RNE) in one
// v_perm_b32: result = { hi16(b+0x8000), hi16(a+0x8000) }.
__device__ __forceinline__ unsigned pack2bf(float a, float b) {
    union { float f; unsigned u; } ua, ub;
    ua.f = a; ub.f = b;
    return __builtin_amdgcn_perm(ub.u + 0x8000u, ua.u + 0x8000u, 0x07060302u);
}

__device__ __forceinline__ __bf16 f2bf(float f) {
    union { float f; unsigned u; } in; in.f = f;
    union { unsigned short s; __bf16 b; } out;
    out.s = (unsigned short)((in.u + 0x8000u) >> 16);
    return out.b;
}

// ---- LayerNorm over one row of E=1024, 256 threads per row -----------------
__device__ __forceinline__ void ln_row(const float* __restrict__ src,
                                       const float* __restrict__ wgt,
                                       const float* __restrict__ bias,
                                       float* __restrict__ dst) {
    __shared__ float red[16];
    const int tid = threadIdx.x;
    float vals[4];
    float s = 0.f, s2 = 0.f;
#pragma unroll
    for (int i = 0; i < 4; ++i) {
        float v = src[tid + i * 256];
        vals[i] = v; s += v; s2 += v * v;
    }
#pragma unroll
    for (int off = 16; off > 0; off >>= 1) {   // wave32 reduce
        s  += __shfl_xor(s,  off);
        s2 += __shfl_xor(s2, off);
    }
    const int wave = tid >> 5, lane = tid & 31;
    if (lane == 0) { red[wave] = s; red[8 + wave] = s2; }
    __syncthreads();
    if (tid == 0) {
        float ts = 0.f, ts2 = 0.f;
        for (int i = 0; i < 8; ++i) { ts += red[i]; ts2 += red[8 + i]; }
        red[0] = ts; red[8] = ts2;
    }
    __syncthreads();
    const float mean = red[0] * (1.f / RWKV_E);
    const float var  = red[8] * (1.f / RWKV_E) - mean * mean;
    const float rs   = rsqrtf(var + 1e-5f);
#pragma unroll
    for (int i = 0; i < 4; ++i) {
        int e = tid + i * 256;
        dst[e] = (vals[i] - mean) * rs * wgt[e] + bias[e];
    }
}

__global__ void k_embed_ln0(const int* __restrict__ tokens,
                            const float* __restrict__ emb,
                            const float* __restrict__ w,
                            const float* __restrict__ b,
                            float* __restrict__ x) {
    const int row = blockIdx.x;
    const int tok = tokens[row];
    ln_row(emb + (size_t)tok * RWKV_E, w, b, x + (size_t)row * RWKV_E);
}

__global__ void k_layernorm(const float* __restrict__ x,
                            const float* __restrict__ w,
                            const float* __restrict__ b,
                            float* __restrict__ xn) {
    const int row = blockIdx.x;
    ln_row(x + (size_t)row * RWKV_E, w, b, xn + (size_t)row * RWKV_E);
}

// Time-shift mixing, vectorized x4.  prev = (t==0) ? state row : xn[t-1].
__global__ void k_mix(const float* __restrict__ xn,
                      const float* __restrict__ st_row_in,   // pre-offset row
                      float* __restrict__ st_row_out,        // pre-offset row
                      const float* __restrict__ tmk,
                      const float* __restrict__ tmv,         // may be null
                      const float* __restrict__ tmr,
                      __bf16* __restrict__ xk,
                      __bf16* __restrict__ xv,               // may be null
                      __bf16* __restrict__ xr,
                      int T, int stateStride) {
    const size_t g  = (size_t)blockIdx.x * blockDim.x + threadIdx.x; // M*E/4
    const int e4 = (int)(g & 255);                // E/4 = 256 float4 per row
    const int t  = (int)((g >> 8) % T);
    const int b  = (int)(g / ((size_t)256 * T));
    const float4 cur = reinterpret_cast<const float4*>(xn)[g];
    float4 prev;
    if (t == 0)
        prev = *reinterpret_cast<const float4*>(st_row_in + (size_t)b * stateStride + e4 * 4);
    else
        prev = reinterpret_cast<const float4*>(xn)[g - 256];
    if (t == T - 1)
        *reinterpret_cast<float4*>(st_row_out + (size_t)b * stateStride + e4 * 4) = cur;

    const float4 mk = reinterpret_cast<const float4*>(tmk)[e4];
    const float4 mr = reinterpret_cast<const float4*>(tmr)[e4];
    uint2 ok, orr;
    ok.x  = pack2bf(cur.x * mk.x + prev.x * (1.f - mk.x),
                    cur.y * mk.y + prev.y * (1.f - mk.y));
    ok.y  = pack2bf(cur.z * mk.z + prev.z * (1.f - mk.z),
                    cur.w * mk.w + prev.w * (1.f - mk.w));
    orr.x = pack2bf(cur.x * mr.x + prev.x * (1.f - mr.x),
                    cur.y * mr.y + prev.y * (1.f - mr.y));
    orr.y = pack2bf(cur.z * mr.z + prev.z * (1.f - mr.z),
                    cur.w * mr.w + prev.w * (1.f - mr.w));
    reinterpret_cast<uint2*>(xk)[g] = ok;
    reinterpret_cast<uint2*>(xr)[g] = orr;
    if (xv) {
        const float4 mv = reinterpret_cast<const float4*>(tmv)[e4];
        uint2 ov;
        ov.x = pack2bf(cur.x * mv.x + prev.x * (1.f - mv.x),
                       cur.y * mv.y + prev.y * (1.f - mv.y));
        ov.y = pack2bf(cur.z * mv.z + prev.z * (1.f - mv.z),
                       cur.w * mv.w + prev.w * (1.f - mv.w));
        reinterpret_cast<uint2*>(xv)[g] = ov;
    }
}

// a_bf = bf16(r_sigmoid * wkv_out), vectorized x4.
__global__ void k_gate(const float* __restrict__ r,
                       const float* __restrict__ o,
                       __bf16* __restrict__ a, size_t n4) {
    const size_t g = (size_t)blockIdx.x * blockDim.x + threadIdx.x;
    if (g >= n4) return;
    const float4 rv = reinterpret_cast<const float4*>(r)[g];
    const float4 ov = reinterpret_cast<const float4*>(o)[g];
    uint2 res;
    res.x = pack2bf(rv.x * ov.x, rv.y * ov.y);
    res.y = pack2bf(rv.z * ov.z, rv.w * ov.w);
    reinterpret_cast<uint2*>(a)[g] = res;
}

// Running-max stabilized WKV recurrence: one thread per (b,e) channel.
__global__ void k_wkv(const float* __restrict__ kbuf,
                      const float* __restrict__ vbuf,
                      const float* __restrict__ tf,
                      const float* __restrict__ td,
                      const float* __restrict__ st_in,   // pre-offset to row 5i+2
                      float* __restrict__ st_out,
                      float* __restrict__ out,
                      int B, int T, int stateStride) {
    const int id = blockIdx.x * blockDim.x + threadIdx.x;
    if (id >= B * RWKV_E) return;
    const int b = id / RWKV_E, e = id % RWKV_E;
    const size_t sb = (size_t)b * stateStride;
    float aa = st_in[sb + 0 * RWKV_E + e];
    float bb = st_in[sb + 1 * RWKV_E + e];
    float pp = st_in[sb + 2 * RWKV_E + e];
    const float u = tf[e], w = td[e];
    const float* kp = kbuf + (size_t)b * T * RWKV_E + e;
    const float* vp = vbuf + (size_t)b * T * RWKV_E + e;
    float*       op = out  + (size_t)b * T * RWKV_E + e;
    for (int t = 0; t < T; ++t) {
        const float kt = kp[(size_t)t * RWKV_E];
        const float vt = vp[(size_t)t * RWKV_E];
        float ww = u + kt;
        float p  = fmaxf(pp, ww);
        float e1 = __expf(pp - p), e2 = __expf(ww - p);
        op[(size_t)t * RWKV_E] = (e1 * aa + e2 * vt) / (e1 * bb + e2);
        ww = pp + w;
        p  = fmaxf(ww, kt);
        e1 = __expf(ww - p); e2 = __expf(kt - p);
        aa = e1 * aa + e2 * vt;
        bb = e1 * bb + e2;
        pp = p;
    }
    st_out[sb + 0 * RWKV_E + e] = aa;
    st_out[sb + 1 * RWKV_E + e] = bb;
    st_out[sb + 2 * RWKV_E + e] = pp;
}

// ---- bf16 WMMA GEMM: C[M,N] = A_bf16[M,K] @ W_f32->bf16[K,N] ---------------
// 256 thr = 8 waves, tile 128x128xBK32, double-buffered LDS (padded stride 40).
// Staging is macro-inlined into NAMED registers (no arrays -> no scratch).
// Epilogue: 0 store f32; 1 sigmoid f32; 2 relu^2 bf16; 3 xacc += gate*C.

#define GEMM_FETCH(K0)                                                         \
    {                                                                          \
        aReg0 = *(reinterpret_cast<const uint4*>(                              \
                      A + (size_t)(m0 + arow) * K + (K0)) + aseg);             \
        aReg1 = *(reinterpret_cast<const uint4*>(                              \
                      A + (size_t)(m0 + arow + 64) * K + (K0)) + aseg);        \
        wv0 = *reinterpret_cast<const float4*>(                                \
            W + (size_t)((K0) + bk4 + 0) * N + n0 + bn4);                      \
        wv1 = *reinterpret_cast<const float4*>(                                \
            W + (size_t)((K0) + bk4 + 1) * N + n0 + bn4);                      \
        wv2 = *reinterpret_cast<const float4*>(                                \
            W + (size_t)((K0) + bk4 + 2) * N + n0 + bn4);                      \
        wv3 = *reinterpret_cast<const float4*>(                                \
            W + (size_t)((K0) + bk4 + 3) * N + n0 + bn4);                      \
    }

#define GEMM_STAGE(BUF)                                                        \
    {                                                                          \
        *reinterpret_cast<uint4*>(lds_a[BUF] + arow * TSTR + aseg * 8) = aReg0;\
        *reinterpret_cast<uint4*>(lds_a[BUF] + (arow + 64) * TSTR + aseg * 8) = aReg1; \
        uint2 p0, p1, p2, p3;                                                  \
        p0.x = pack2bf(wv0.x, wv1.x); p0.y = pack2bf(wv2.x, wv3.x);            \
        p1.x = pack2bf(wv0.y, wv1.y); p1.y = pack2bf(wv2.y, wv3.y);            \
        p2.x = pack2bf(wv0.z, wv1.z); p2.y = pack2bf(wv2.z, wv3.z);            \
        p3.x = pack2bf(wv0.w, wv1.w); p3.y = pack2bf(wv2.w, wv3.w);            \
        *reinterpret_cast<uint2*>(lds_b[BUF] + (bn4 + 0) * TSTR + bk4) = p0;   \
        *reinterpret_cast<uint2*>(lds_b[BUF] + (bn4 + 1) * TSTR + bk4) = p1;   \
        *reinterpret_cast<uint2*>(lds_b[BUF] + (bn4 + 2) * TSTR + bk4) = p2;   \
        *reinterpret_cast<uint2*>(lds_b[BUF] + (bn4 + 3) * TSTR + bk4) = p3;   \
    }

__global__ void __launch_bounds__(256)
k_gemm_bf16(const __bf16* __restrict__ A, const float* __restrict__ W,
            int M, int N, int K, int mode,
            float* __restrict__ outF, __bf16* __restrict__ outB,
            const float* __restrict__ gate, float* __restrict__ xacc) {
    __shared__ __bf16 lds_a[2][BM * TSTR];
    __shared__ __bf16 lds_b[2][BN * TSTR];

    const int tid  = threadIdx.x;
    const int m0   = blockIdx.y * BM;
    const int n0   = blockIdx.x * BN;
    const int wave = tid >> 5, lane = tid & 31;
    const int wm = wave >> 1, wn = wave & 1;
    const int lrow  = lane & 15;
    const int khalf = (lane >> 4) << 3;         // 0 or 8
    const int arow = tid >> 2, aseg = tid & 3;  // A staging slot
    const int bn4 = (tid & 31) << 2;            // B staging: 4 cols
    const int bk4 = (tid >> 5) << 2;            // B staging: 4 k-rows

    v8f acc[2][4];
#pragma unroll
    for (int mi = 0; mi < 2; ++mi)
#pragma unroll
        for (int ni = 0; ni < 4; ++ni)
#pragma unroll
            for (int q = 0; q < 8; ++q) acc[mi][ni][q] = 0.f;

    uint4 aReg0, aReg1;
    float4 wv0, wv1, wv2, wv3;

    GEMM_FETCH(0)
    GEMM_STAGE(0)
    __syncthreads();

    int buf = 0;
    for (int k0 = 0; k0 < K; k0 += BK) {
        const bool nxt = (k0 + BK) < K;
        if (nxt) GEMM_FETCH(k0 + BK)
        if (k0 + 2 * BK < K) {   // L2 prefetch two steps ahead
            __builtin_prefetch(W + (size_t)(k0 + 2 * BK + bk4) * N + n0 + bn4, 0, 1);
            __builtin_prefetch(A + (size_t)(m0 + (tid >> 1)) * K + k0 + 2 * BK, 0, 1);
        }

        // Fragments: ISA 16-bit A 16x32 layout -> lane row (lane&15),
        // K [khalf,khalf+8) then [khalf+16,khalf+24): two b128 LDS loads each.
        v16bf af[2], bfr[4];
#pragma unroll
        for (int mi = 0; mi < 2; ++mi) {
            const __bf16* p = lds_a[buf] + (wm * 32 + mi * 16 + lrow) * TSTR + khalf;
            v8bf* h = reinterpret_cast<v8bf*>(&af[mi]);
            h[0] = reinterpret_cast<const v8bf*>(p)[0];
            h[1] = reinterpret_cast<const v8bf*>(p)[2];   // +16 elements
        }
#pragma unroll
        for (int ni = 0; ni < 4; ++ni) {
            const __bf16* p = lds_b[buf] + (wn * 64 + ni * 16 + lrow) * TSTR + khalf;
            v8bf* h = reinterpret_cast<v8bf*>(&bfr[ni]);
            h[0] = reinterpret_cast<const v8bf*>(p)[0];
            h[1] = reinterpret_cast<const v8bf*>(p)[2];
        }
#pragma unroll
        for (int mi = 0; mi < 2; ++mi)
#pragma unroll
            for (int ni = 0; ni < 4; ++ni)
                acc[mi][ni] = __builtin_amdgcn_wmma_f32_16x16x32_bf16(
                    false, af[mi], false, bfr[ni], (short)0, acc[mi][ni],
                    false, false);

        if (nxt) GEMM_STAGE(buf ^ 1)
        __syncthreads();
        buf ^= 1;
    }

    // C/D layout: lanes 0-15: n=lane, m=q; lanes 16-31: n=lane-16, m=q+8.
#pragma unroll
    for (int mi = 0; mi < 2; ++mi) {
#pragma unroll
        for (int ni = 0; ni < 4; ++ni) {
            const int rbase = m0 + wm * 32 + mi * 16 + ((lane >> 4) << 3);
            const int c     = n0 + wn * 64 + ni * 16 + lrow;
#pragma unroll
            for (int q = 0; q < 8; ++q) {
                const size_t o  = (size_t)(rbase + q) * N + c;
                const float  cv = acc[mi][ni][q];
                if (mode == 0) {
                    outF[o] = cv;
                } else if (mode == 1) {
                    outF[o] = 1.f / (1.f + __expf(-cv));
                } else if (mode == 2) {
                    const float t = fmaxf(cv, 0.f);
                    outB[o] = f2bf(t * t);
                } else {
                    const float g = gate ? gate[o] : 1.f;
                    xacc[o] += g * cv;
                }
            }
        }
    }
}

// ---------------------------------------------------------------------------
extern "C" void kernel_launch(void* const* d_in, const int* in_sizes, int n_in,
                              void* d_out, int out_size, void* d_ws, size_t ws_size,
                              hipStream_t stream) {
    (void)in_sizes; (void)n_in; (void)out_size; (void)ws_size;
    const int B = 8, T = 512, E = 1024, L = 6, H = 4096;
    const int M = B * T;                 // 4096 rows
    const int stateStride = 5 * L * E;   // 30720

    const int*   tokens  = (const int*)  d_in[0];
    const float* stateIn = (const float*)d_in[1];
    const float* emb     = (const float*)d_in[2];
    const float* ln0w    = (const float*)d_in[3];
    const float* ln0b    = (const float*)d_in[4];
    const float* ln1w    = (const float*)d_in[5];
    const float* ln1b    = (const float*)d_in[6];
    const float* ln2w    = (const float*)d_in[7];
    const float* ln2b    = (const float*)d_in[8];
    const float* att_tmk = (const float*)d_in[9];
    const float* att_tmv = (const float*)d_in[10];
    const float* att_tmr = (const float*)d_in[11];
    const float* tf      = (const float*)d_in[12];
    const float* td      = (const float*)d_in[13];
    const float* att_kw  = (const float*)d_in[14];
    const float* att_vw  = (const float*)d_in[15];
    const float* att_rw  = (const float*)d_in[16];
    const float* att_ow  = (const float*)d_in[17];
    const float* ffn_tmk = (const float*)d_in[18];
    const float* ffn_tmr = (const float*)d_in[19];
    const float* ffn_kw  = (const float*)d_in[20];
    const float* ffn_vw  = (const float*)d_in[21];
    const float* ffn_rw  = (const float*)d_in[22];
    float* stateOut = (float*)d_out;

    // ---- workspace carve-out (~180 MB total) ----
    char* ws = (char*)d_ws;
    size_t off = 0;
    auto carve = [&](size_t bytes) -> void* {
        void* p = ws + off;
        off += (bytes + 255) & ~(size_t)255;
        return p;
    };
    const size_t ME = (size_t)M * E;     // 4M elements
    const size_t MH = (size_t)M * H;     // 16M elements
    float*  x      = (float*) carve(ME * 4);
    float*  xn     = (float*) carve(ME * 4);
    float*  kbuf   = (float*) carve(ME * 4);
    float*  vbuf   = (float*) carve(ME * 4);
    float*  rbuf   = (float*) carve(ME * 4);
    float*  wkvout = (float*) carve(ME * 4);
    float*  rffn   = (float*) carve(ME * 4);
    __bf16* xk_bf  = (__bf16*)carve(ME * 2);
    __bf16* xv_bf  = (__bf16*)carve(ME * 2);
    __bf16* xr_bf  = (__bf16*)carve(ME * 2);
    __bf16* abuf   = (__bf16*)carve(ME * 2);
    __bf16* hbuf   = (__bf16*)carve(MH * 2);

    const dim3 blk(256);
    const dim3 gRows(M);                          // one block per row (LN)
    const dim3 gElem4((unsigned)(ME / 4 / 256));  // x4-vectorized elementwise
    const dim3 gWkv((unsigned)((B * E + 255) / 256));
    const dim3 gGemmE(E / BN, M / BM);            // N=1024
    const dim3 gGemmH(H / BN, M / BM);            // N=4096

    k_embed_ln0<<<gRows, blk, 0, stream>>>(tokens, emb, ln0w, ln0b, x);

    for (int i = 0; i < L; ++i) {
        const size_t iE  = (size_t)i * E;
        const size_t iEE = (size_t)i * E * E;
        const size_t iEH = (size_t)i * E * H;

        // ===== time mixing =====
        k_layernorm<<<gRows, blk, 0, stream>>>(x, ln1w + iE, ln1b + iE, xn);
        k_mix<<<gElem4, blk, 0, stream>>>(
            xn, stateIn + (size_t)(5 * i + 1) * E, stateOut + (size_t)(5 * i + 1) * E,
            att_tmk + iE, att_tmv + iE, att_tmr + iE,
            xk_bf, xv_bf, xr_bf, T, stateStride);
        k_gemm_bf16<<<gGemmE, blk, 0, stream>>>(xk_bf, att_kw + iEE, M, E, E, 0,
                                                kbuf, nullptr, nullptr, nullptr);
        k_gemm_bf16<<<gGemmE, blk, 0, stream>>>(xv_bf, att_vw + iEE, M, E, E, 0,
                                                vbuf, nullptr, nullptr, nullptr);
        k_gemm_bf16<<<gGemmE, blk, 0, stream>>>(xr_bf, att_rw + iEE, M, E, E, 1,
                                                rbuf, nullptr, nullptr, nullptr);
        k_wkv<<<gWkv, blk, 0, stream>>>(kbuf, vbuf, tf + iE, td + iE,
                                        stateIn + (size_t)(5 * i + 2) * E,
                                        stateOut + (size_t)(5 * i + 2) * E,
                                        wkvout, B, T, stateStride);
        k_gate<<<gElem4, blk, 0, stream>>>(rbuf, wkvout, abuf, ME / 4);
        k_gemm_bf16<<<gGemmE, blk, 0, stream>>>(abuf, att_ow + iEE, M, E, E, 3,
                                                nullptr, nullptr, nullptr, x);

        // ===== channel mixing =====
        k_layernorm<<<gRows, blk, 0, stream>>>(x, ln2w + iE, ln2b + iE, xn);
        k_mix<<<gElem4, blk, 0, stream>>>(
            xn, stateIn + (size_t)(5 * i + 0) * E, stateOut + (size_t)(5 * i + 0) * E,
            ffn_tmk + iE, nullptr, ffn_tmr + iE,
            xk_bf, nullptr, xr_bf, T, stateStride);
        k_gemm_bf16<<<gGemmH, blk, 0, stream>>>(xk_bf, ffn_kw + iEH, M, H, E, 2,
                                                nullptr, hbuf, nullptr, nullptr);
        k_gemm_bf16<<<gGemmE, blk, 0, stream>>>(xr_bf, ffn_rw + iEE, M, E, E, 1,
                                                rffn, nullptr, nullptr, nullptr);
        k_gemm_bf16<<<gGemmE, blk, 0, stream>>>(hbuf, ffn_vw + iEH, M, E, H, 3,
                                                nullptr, nullptr, rffn, x);
    }
}
